// PRCD_MAP_NAM_30176440221721
// MI455X (gfx1250) — matrix operand order repaired and checked by reading, not compiled, not run on hardware
//
#include <hip/hip_runtime.h>

// CDNA5 / gfx1250 WMMA types
typedef __attribute__((ext_vector_type(16))) _Float16 v16h;
typedef __attribute__((ext_vector_type(8)))  float    v8f;

#define D_ 64
#define T_ 2048
#define K_ 2
#define H_ 16
#define E_ (D_ * (D_ - 1))

// One wave32 per block. Block (blockIdx.x = t-tile, blockIdx.y = d) produces
// pred[tb*16 .. tb*16+15, d] by accumulating the 63 incoming edges (src i -> d)
// in registers (deterministic segment-sum), using one
// V_WMMA_F32_16X16X32_F16 per edge for the (16x16)@(16x16) hidden GEMM
// (K zero-padded 16 -> 32, f32 accumulation).
__global__ __launch_bounds__(32)
void PRCD_edge_mlp_wmma(const float* __restrict__ X_t,    // [T, D]
                        const float* __restrict__ X_lags, // [K, T, D]
                        const float* __restrict__ W1,     // [E, H]
                        const float* __restrict__ b1,     // [E, H]
                        const float* __restrict__ W2,     // [E, H, H]
                        const float* __restrict__ b2,     // [E, H]
                        const float* __restrict__ W3,     // [E, H]
                        const float* __restrict__ b3,     // [E]
                        const float* __restrict__ Wk,     // [K, D, D]
                        float* __restrict__ pred)         // [T, D]
{
    const int lane  = threadIdx.x;      // 0..31
    const int half  = lane >> 4;        // 0 or 1  (lane group)
    const int lmod  = lane & 15;
    const int tbase = blockIdx.x * 16;  // t tile base
    const int d     = blockIdx.y;       // destination node

    // A-matrix: this lane holds row M = lmod  ->  t = tbase + lmod
    const int tA = tbase + lmod;

    v8f   part  = {0.f, 0.f, 0.f, 0.f, 0.f, 0.f, 0.f, 0.f};
    float b3sum = 0.f;

    // Edges with dst == d : src i != d, edge id e = i*63 + d - (d > i).
    // (Row-major enumeration of off-diagonal (i,j) pairs, as in setup_inputs.)
    for (int i = 0; i < D_; ++i) {
        if (i == d) continue;                      // uniform branch: EXEC stays full
        const int e = i * 63 + d - (d > i ? 1 : 0);

        // Prefetch next edge's W2 tile (1 KB) -> global_prefetch_b8
        {
            int inext = i + 1;
            if (inext == d) ++inext;
            if (inext < D_) {
                const int en = inext * 63 + d - (d > inext ? 1 : 0);
                __builtin_prefetch(W2 + (size_t)en * (H_ * H_), 0, 1);
            }
        }

        // --- layer 1: h1[t, h] = relu(X_t[t, i] * W1[e, h] + b1[e, h]) ---
        // A-matrix f16 16x32 layout: lane holds M = lmod;
        //   element idx 0..7  -> K = idx + 8*half  (real, K < 16)
        //   element idx 8..15 -> K >= 16           (zero pad)
        const float xv   = X_t[tA * D_ + i];
        const float* w1p = W1 + e * H_ + half * 8;
        const float* b1p = b1 + e * H_ + half * 8;
        v16h a;
        #pragma unroll
        for (int idx = 0; idx < 8; ++idx) {
            float h1v = fmaf(xv, w1p[idx], b1p[idx]);
            h1v = h1v > 0.f ? h1v : 0.f;
            a[idx]     = (_Float16)h1v;
            a[idx + 8] = (_Float16)0.f;
        }

        // --- B = W2[e] (16x16), padded to 32x16 ---
        // B-matrix layout: lane holds column N = lmod;
        //   element idx 0..7  -> K = idx + 8*half  (real rows 0..15)
        //   element idx 8..15 -> K >= 16           (zero pad)
        const float* w2p = W2 + (size_t)e * (H_ * H_) + lmod;
        v16h b;
        #pragma unroll
        for (int idx = 0; idx < 16; ++idx) {
            float w2v = (idx < 8) ? w2p[(idx + 8 * half) * H_] : 0.f;
            b[idx] = (_Float16)w2v;
        }

        // --- h2pre = h1 @ W2[e]  (one WMMA, f32 accumulate) ---
        v8f c = {0.f, 0.f, 0.f, 0.f, 0.f, 0.f, 0.f, 0.f};
        v8f h2 = __builtin_amdgcn_wmma_f32_16x16x32_f16(
            /*neg_a=*/false, a, /*neg_b=*/false, b,
            /*c_mod=*/(short)0, c, /*reuse_a=*/false, /*reuse_b=*/false);

        // D layout: VGPR r, lane -> (t = tbase + r + 8*half, g = lmod)
        // --- layer 2 bias+relu, fold W3 dot into per-lane partial ---
        const float b2v = b2[e * H_ + lmod];
        const float w3v = W3[e * H_ + lmod];
        #pragma unroll
        for (int r = 0; r < 8; ++r) {
            float hv = h2[r] + b2v;
            hv = hv > 0.f ? hv : 0.f;
            part[r] += hv * w3v;
        }
        b3sum += b3[e];
    }

    // Reduce over g (the 16 lanes of each half) and store with lag term.
    #pragma unroll
    for (int r = 0; r < 8; ++r) {
        float v = part[r];
        v += __shfl_xor(v, 1, 32);
        v += __shfl_xor(v, 2, 32);
        v += __shfl_xor(v, 4, 32);
        v += __shfl_xor(v, 8, 32);
        if (lmod == 0) {
            const int t = tbase + r + 8 * half;
            float lag = 0.f;
            #pragma unroll
            for (int k = 0; k < K_; ++k) {
                const float* xl = X_lags + ((size_t)k * T_ + t) * D_;
                const float* wk = Wk + (size_t)k * D_ * D_ + d;
                #pragma unroll 8
                for (int c2 = 0; c2 < D_; ++c2)
                    lag = fmaf(xl[c2], wk[c2 * D_], lag);
            }
            pred[t * D_ + d] = v + b3sum + lag;
        }
    }
}

extern "C" void kernel_launch(void* const* d_in, const int* in_sizes, int n_in,
                              void* d_out, int out_size, void* d_ws, size_t ws_size,
                              hipStream_t stream) {
    const float* X_t    = (const float*)d_in[0];
    const float* X_lags = (const float*)d_in[1];
    const float* W1     = (const float*)d_in[2];
    const float* b1     = (const float*)d_in[3];
    const float* W2     = (const float*)d_in[4];
    const float* b2     = (const float*)d_in[5];
    const float* W3     = (const float*)d_in[6];
    const float* b3     = (const float*)d_in[7];
    const float* Wk     = (const float*)d_in[8];
    // d_in[9] = src, d_in[10] = dst: the fixed row-major off-diagonal
    // enumeration from setup_inputs; encoded analytically in the kernel.
    (void)in_sizes; (void)n_in; (void)d_ws; (void)ws_size; (void)out_size;

    float* pred = (float*)d_out;
    dim3 grid(T_ / 16, D_);
    PRCD_edge_mlp_wmma<<<grid, dim3(32), 0, stream>>>(
        X_t, X_lags, W1, b1, W2, b2, W3, b3, Wk, pred);
}
